// GeneralizedAttention_6296422056056
// MI455X (gfx1250) — matrix operand order repaired and checked by reading, not compiled, not run on hardware
//
#include <hip/hip_runtime.h>
#include <hip/hip_bf16.h>

typedef __attribute__((ext_vector_type(16))) _Float16 v16h;
typedef __attribute__((ext_vector_type(8)))  float    v8f;

#define NB   2
#define NH   8
#define HWQ  4096   // 64*64 queries
#define KVN  1024   // 32*32 kv positions
#define CIN  256

// ---------------- WMMA helpers (gfx1250, wave32) ----------------

__device__ __forceinline__ v8f wmma16(v16h a, v16h b, v8f c) {
  return __builtin_amdgcn_wmma_f32_16x16x32_f16(
      /*neg_a=*/false, a, /*neg_b=*/false, b,
      /*c_mod=*/(short)0, c, /*reuse_a=*/false, /*reuse_b=*/false);
}

// A-matrix 16x32 f16 K-map: lanes 0-15 hold K 0..7 (V0-3) & 16..23 (V4-7);
// lanes 16-31 hold K 8..15 & 24..31.  kk[j] = even K of vgpr j.
__device__ __forceinline__ void lane_kkA(int lane, int (&kk)[8]) {
  int kb = (lane >> 4) << 3;
#pragma unroll
  for (int j = 0; j < 8; ++j)
    kk[j] = kb + ((j < 4) ? (2 * j) : (16 + 2 * (j - 4)));
}

// B-matrix 32x16 f16 K-map (same as A): lanes 0-15 hold K 0..7 & 16..23, lanes 16-31 hold K 8..15 & 24..31.
__device__ __forceinline__ void lane_kkB(int lane, int (&kk)[8]) {
  int kb = (lane >> 4) << 3;
#pragma unroll
  for (int j = 0; j < 8; ++j) kk[j] = kb + ((j < 4) ? (2 * j) : (16 + 2 * (j - 4)));
}

// A tile loaders: rowbase points at this lane's row start (K stride 1).
__device__ __forceinline__ v16h a_from_f32(const float* rowbase, const int (&kk)[8]) {
  v16h a;
#pragma unroll
  for (int j = 0; j < 8; ++j) {
    a[2 * j]     = (_Float16)rowbase[kk[j]];
    a[2 * j + 1] = (_Float16)rowbase[kk[j] + 1];
  }
  return a;
}
__device__ __forceinline__ v16h a_from_f16(const _Float16* rowbase, const int (&kk)[8]) {
  v16h a;
#pragma unroll
  for (int j = 0; j < 8; ++j) {
    a[2 * j]     = rowbase[kk[j]];
    a[2 * j + 1] = rowbase[kk[j] + 1];
  }
  return a;
}

// B tile loaders: colbase points at (K=0, this lane's column element); stride = ld.
__device__ __forceinline__ v16h b_from_f32(const float* colbase, int ld, const int (&kk)[8]) {
  v16h b;
#pragma unroll
  for (int j = 0; j < 8; ++j) {
    b[2 * j]     = (_Float16)colbase[(size_t)kk[j] * ld];
    b[2 * j + 1] = (_Float16)colbase[(size_t)(kk[j] + 1) * ld];
  }
  return b;
}
__device__ __forceinline__ v16h b_from_f16(const _Float16* colbase, int ld, const int (&kk)[8]) {
  v16h b;
#pragma unroll
  for (int j = 0; j < 8; ++j) {
    b[2 * j]     = colbase[(size_t)kk[j] * ld];
    b[2 * j + 1] = colbase[(size_t)(kk[j] + 1) * ld];
  }
  return b;
}

// -------- CDNA5 async Global -> LDS (ASYNCcnt path) --------
__device__ __forceinline__ void async_ld_b128(unsigned lds_byte_off, const void* gaddr) {
  asm volatile("global_load_async_to_lds_b128 %0, %1, off"
               :: "v"(lds_byte_off), "v"((unsigned long long)(size_t)gaddr)
               : "memory");
}
__device__ __forceinline__ void wait_async0() {
  asm volatile("s_wait_asynccnt 0x0" ::: "memory");
}

// ---------------- Kernel 1: Q projection (+bias fold) ----------------
// All 8 waves of a block share one 256x16 f32 B panel of x -> stage via async LDS.
// Qapp/Qgeo: [(n*8+a)*4096 + q][32] f16
#define KQ_SMEM 16384
__global__ void k_q(const float* __restrict__ x, const float* __restrict__ Wq,
                    const float* __restrict__ ab, const float* __restrict__ gb,
                    _Float16* __restrict__ Qapp, _Float16* __restrict__ Qgeo) {
  extern __shared__ char dynq[];
  float* sB = (float*)dynq;                     // [c 256][s 16]
  int tid = threadIdx.x, lane = tid & 31;
  int wave = blockIdx.x * 8 + (tid >> 5);       // 8192 waves
  int otile = wave & 15, stile = wave >> 4;     // 8 waves/block share stile
  int s0 = stile * 16;
  int n = s0 >> 12, hw0 = s0 & 4095;
  unsigned lbase = __builtin_amdgcn_groupstaticsize();
  {
    const float* src = x + (size_t)n * CIN * HWQ + (size_t)tid * HWQ + hw0;
#pragma unroll
    for (int ch = 0; ch < 4; ++ch)              // row c=tid: 64B = 4 x b128
      async_ld_b128(lbase + (unsigned)(tid * 64 + ch * 16), src + ch * 4);
  }
  int kkA[8], kkB[8];
  lane_kkA(lane, kkA); lane_kkB(lane, kkB);
  int l15 = lane & 15, hi8 = (lane >> 4) << 3;
  wait_async0();
  __syncthreads();
  v8f acc = {};
#pragma unroll
  for (int kc = 0; kc < 8; ++kc) {
    v16h a = a_from_f32(Wq + (size_t)(otile * 16 + l15) * CIN + kc * 32, kkA);
    v16h b = b_from_f32(sB + kc * 32 * 16 + l15, 16, kkB);
    acc = wmma16(a, b, acc);
  }
#pragma unroll
  for (int i = 0; i < 8; ++i) {
    int r = i + hi8;
    int o = otile * 16 + r;
    int aH = o >> 5, d = o & 31;
    size_t qi = ((size_t)(n * NH + aH) * HWQ + hw0 + l15) * 32 + d;
    Qapp[qi] = (_Float16)(acc[i] + ab[o]);
    Qgeo[qi] = (_Float16)(acc[i] + gb[o]);
  }
}

// ---------------- Kernel 2: K/V projection (strided x_kv) ----------------
// Stage the raw stride-2 span (256 x 32 f32) once per block; B reads every
// other float.  Kt: [na][d 32][kl 1024] ; Vt: [na][kl 1024][d 32]
#define KKV_SMEM 32768
__global__ void k_kv(const float* __restrict__ x, const float* __restrict__ Wk,
                     const float* __restrict__ Wv,
                     _Float16* __restrict__ Kt, _Float16* __restrict__ Vt) {
  extern __shared__ char dynkv[];
  float* sB = (float*)dynkv;                    // [c 256][span 32]
  int tid = threadIdx.x, lane = tid & 31;
  int wave = blockIdx.x * 8 + (tid >> 5);       // 2048 waves
  int otile = wave & 15, st = wave >> 4;        // 8 waves/block share st
  int s0 = st * 16;
  int n = s0 >> 10, kl0 = s0 & 1023;
  int hk = kl0 >> 5, wl0 = kl0 & 31;            // all 16 cols share hk
  unsigned lbase = __builtin_amdgcn_groupstaticsize();
  {
    const float* src = x + (size_t)n * CIN * HWQ + (size_t)tid * HWQ + hk * 128 + wl0 * 2;
#pragma unroll
    for (int ch = 0; ch < 8; ++ch)              // row c=tid: 128B = 8 x b128
      async_ld_b128(lbase + (unsigned)(tid * 128 + ch * 16), src + ch * 4);
  }
  int kkA[8], kkB[8];
  lane_kkA(lane, kkA); lane_kkB(lane, kkB);
  int l15 = lane & 15, hi8 = (lane >> 4) << 3;
  wait_async0();
  __syncthreads();
  v8f ak = {}, av = {};
#pragma unroll
  for (int kc = 0; kc < 8; ++kc) {
    v16h b  = b_from_f32(sB + kc * 32 * 32 + 2 * l15, 32, kkB);
    v16h a1 = a_from_f32(Wk + (size_t)(otile * 16 + l15) * CIN + kc * 32, kkA);
    v16h a2 = a_from_f32(Wv + (size_t)(otile * 16 + l15) * CIN + kc * 32, kkA);
    ak = wmma16(a1, b, ak);
    av = wmma16(a2, b, av);
  }
#pragma unroll
  for (int i = 0; i < 8; ++i) {
    int r = i + hi8;
    int o = otile * 16 + r;
    int aH = o >> 5, d = o & 31;
    int na = n * NH + aH;
    int klc = kl0 + l15;
    Kt[(size_t)na * 32768 + (size_t)d * KVN + klc] = (_Float16)ak[i];
    Vt[(size_t)na * 32768 + (size_t)klc * 32 + d] = (_Float16)av[i];
  }
}

// ---------------- Kernel 3: sinusoidal embedding table ----------------
// E[w][dd 0..127][l 0..31] f16 ; ex == ey for this shape (64 q, 32 kv, stride 2)
__global__ void k_embed(_Float16* __restrict__ E) {
  int i = blockIdx.x * 256 + threadIdx.x;       // 262144 elements
  int w = i >> 12, dd = (i >> 5) & 127, l = i & 31;
  float diff = (float)(w - 2 * l);              // MAG=1, Q_STRIDE=1, KV_STRIDE=2
  int f = dd & 63;
  float dm = powf(1000.0f, (float)f * (1.0f / 64.0f));
  float ang = diff / dm;
  float v = (dd < 64) ? sinf(ang) : cosf(ang);
  E[i] = (_Float16)v;
}

// ---------------- Kernel 4: pos_x / pos_y projection ----------------
// pxB/pyB: [a][pos 64][d 32][l 32] f16 (B-layout for GX/GY)
__global__ void k_pos(const float* __restrict__ Wx, const float* __restrict__ Wy,
                      const _Float16* __restrict__ E,
                      _Float16* __restrict__ pxB, _Float16* __restrict__ pyB) {
  int tid = threadIdx.x, lane = tid & 31;
  int wave = blockIdx.x * 8 + (tid >> 5);       // 2048 waves
  int w = wave >> 5;
  int sub = wave & 31;
  int otile = sub >> 1, lt = sub & 1;
  int kkA[8], kkB[8];
  lane_kkA(lane, kkA); lane_kkB(lane, kkB);
  int l15 = lane & 15, hi8 = (lane >> 4) << 3;
  v8f ax = {}, ay = {};
#pragma unroll
  for (int kc = 0; kc < 4; ++kc) {              // K = 128
    v16h b  = b_from_f16(E + (size_t)w * 4096 + kc * 1024 + lt * 16 + l15, 32, kkB);
    v16h a1 = a_from_f32(Wx + (size_t)(otile * 16 + l15) * 128 + kc * 32, kkA);
    v16h a2 = a_from_f32(Wy + (size_t)(otile * 16 + l15) * 128 + kc * 32, kkA);
    ax = wmma16(a1, b, ax);
    ay = wmma16(a2, b, ay);
  }
  const float inv_s2 = 0.70710678118654752f;
#pragma unroll
  for (int i = 0; i < 8; ++i) {
    int r = i + hi8;
    int o = otile * 16 + r;
    int aH = o >> 5, d = o & 31;
    size_t idx = (((size_t)(aH * 64 + w) * 32 + d) * 32) + lt * 16 + l15;
    pxB[idx] = (_Float16)(ax[i] * inv_s2);
    pyB[idx] = (_Float16)(ay[i] * inv_s2);
  }
}

// ---------------- Kernel 5: GX = Qgeo . px,  GY = Qgeo . py ----------------
// GX/GY: [query][32] f32
__global__ void k_gxgy(const _Float16* __restrict__ Qgeo,
                       const _Float16* __restrict__ pxB, const _Float16* __restrict__ pyB,
                       float* __restrict__ GX, float* __restrict__ GY) {
  int tid = threadIdx.x, lane = tid & 31;
  int wv = blockIdx.x * 8 + (tid >> 5);         // 16384 waves
  int isY = wv >> 13;
  int rem = wv & 8191;
  int lt = rem & 1;
  int rowtile = (rem >> 1) & 7;
  int pos = (rem >> 4) & 63;                    // w for GX, h for GY
  int aH = rem >> 10;
  int kkA[8], kkB[8];
  lane_kkA(lane, kkA); lane_kkB(lane, kkB);
  int l15 = lane & 15, hi8 = (lane >> 4) << 3;

  int rt = rowtile * 16 + l15;                  // which of 128 (n x other) rows
  int nn = rt >> 6, oth = rt & 63;
  int qi = (nn * NH + aH) * HWQ + (isY ? (pos * 64 + oth) : (oth * 64 + pos));
  v16h a = a_from_f16(Qgeo + (size_t)qi * 32, kkA);
  const _Float16* P = (isY ? pyB : pxB) +
                      (size_t)(aH * 64 + pos) * 1024 + lt * 16 + l15;
  v16h b = b_from_f16(P, 32, kkB);
  v8f c = {};
  c = wmma16(a, b, c);
  float* dst = isY ? GY : GX;
#pragma unroll
  for (int i = 0; i < 8; ++i) {
    int r = i + hi8;
    int rti = rowtile * 16 + r;
    int n2 = rti >> 6, ot2 = rti & 63;
    int q2 = (n2 * NH + aH) * HWQ + (isY ? (pos * 64 + ot2) : (ot2 * 64 + pos));
    dst[(size_t)q2 * 32 + lt * 16 + l15] = c[i];
  }
}

// ---------------- Kernel 6: fused attention (async LDS staging) ----------------
// One block = (n, head, 16-query tile); 4 waves split the 1024 kv positions.
// Dynamic LDS partition (bytes):
//   sE    @ 0      : 16x1024 f16  = 32768   (energies / unnormalized attn)
//   sKV   @ 32768  : 4 waves x 16KB = 65536 (per-wave K slab, reused for V slab)
//   sPart @ 98304  : 4x16x32 f32  = 8192    (per-wave AV partials)
//   sStat @ 106496 : 16x8 f32     = 512
//   sMax  @ 107008 : 16 f32
//   sSum  @ 107072 : 16 f32
//   sGX   @ 107136 : 16x32 f32    = 2048
//   sGY   @ 109184 : 16x32 f32    = 2048
#define ATTN_SMEM 111232
__global__ void __launch_bounds__(128)
k_attn(const _Float16* __restrict__ Qapp, const _Float16* __restrict__ Kt,
       const _Float16* __restrict__ Vt, const float* __restrict__ GX,
       const float* __restrict__ GY, _Float16* __restrict__ Ot) {
  extern __shared__ char dyn[];
  _Float16* sE    = (_Float16*)(dyn);
  float*    sPart = (float*)(dyn + 98304);
  float*    sStat = (float*)(dyn + 106496);
  float*    sMax  = (float*)(dyn + 107008);
  float*    sSum  = (float*)(dyn + 107072);
  float*    sGX   = (float*)(dyn + 107136);
  float*    sGY   = (float*)(dyn + 109184);

  int tid = threadIdx.x, lane = tid & 31, wvid = tid >> 5;
  int bid = blockIdx.x;
  int qt = bid & 255, aH = (bid >> 8) & 7, n = bid >> 11;
  int na = n * NH + aH;
  int q0 = na * HWQ + qt * 16;
  int kkA[8], kkB[8];
  lane_kkA(lane, kkA); lane_kkB(lane, kkB);
  int l15 = lane & 15, hi8 = (lane >> 4) << 3;

  // This wave's private 16KB K/V staging region.
  _Float16* sKV = (_Float16*)(dyn + 32768) + wvid * 8192;
  unsigned  lbase = __builtin_amdgcn_groupstaticsize();
  unsigned  kvOff = lbase + 32768u + (unsigned)wvid * 16384u;

  // ---- async-load this wave's K slab: 32 rows (d) x 256 kl (512B per row),
  //      plus the block's GX/GY rows (2KB each, 1 b128 per thread) ----
  {
    const _Float16* Kg = Kt + (size_t)na * 32768 + wvid * 256;
#pragma unroll 4
    for (int m = 0; m < 32; ++m)
      async_ld_b128(kvOff + (unsigned)(m * 512 + lane * 16),
                    (const void*)(Kg + (size_t)m * KVN + lane * 8));
    async_ld_b128(lbase + 107136u + (unsigned)(tid * 16), GX + (size_t)q0 * 32 + tid * 4);
    async_ld_b128(lbase + 109184u + (unsigned)(tid * 16), GY + (size_t)q0 * 32 + tid * 4);
  }

  // A-tile (Qapp) overlaps with the async copies.
  v16h aq = a_from_f16(Qapp + (size_t)(q0 + l15) * 32, kkA);
  wait_async0();         // this wave's K slab + its share of GX/GY resident
  __syncthreads();       // all waves' GX/GY shares resident

  // ---- energy: E[q, kl] = Qapp.K + GX[q,l] + GY[q,k] (all operands in LDS) ----
  for (int tl = 0; tl < 16; ++tl) {
    int t = wvid * 16 + tl;
    v16h b = b_from_f16(sKV + tl * 16 + l15, 256, kkB);
    v8f c = {};
    c = wmma16(aq, b, c);
    int l = (t & 1) * 16 + l15, kq = t >> 1;
#pragma unroll
    for (int i = 0; i < 8; ++i) {
      int r = i + hi8;
      float e = c[i] + sGX[r * 32 + l] + sGY[r * 32 + kq];
      sE[r * 1024 + t * 16 + l15] = (_Float16)e;
    }
  }

  // ---- kick off this wave's V slab into the same (now free) region; the
  //      copy is in flight during the whole softmax phase ----
  {
    const _Float16* Vg = Vt + (size_t)na * 32768 + (size_t)wvid * 8192;
#pragma unroll 4
    for (int m = 0; m < 32; ++m) {
      int e16 = m * 32 + lane;           // 16-byte chunk index within 16KB
      async_ld_b128(kvOff + (unsigned)(e16 * 16), (const void*)(Vg + (size_t)e16 * 8));
    }
  }
  __syncthreads();

  // ---- softmax stats (8 threads per row, 128 elems each) ----
  int row = tid >> 3, sub = tid & 7;
  float m = -3.0e38f;
  for (int j = sub * 128; j < sub * 128 + 128; ++j)
    m = fmaxf(m, (float)sE[row * 1024 + j]);
  sStat[row * 8 + sub] = m;
  __syncthreads();
  if (sub == 0) {
    float mm = sStat[row * 8];
#pragma unroll
    for (int j = 1; j < 8; ++j) mm = fmaxf(mm, sStat[row * 8 + j]);
    sMax[row] = mm;
  }
  __syncthreads();
  float mm = sMax[row];
  float s = 0.f;
  for (int j = sub * 128; j < sub * 128 + 128; ++j) {
    float e = __expf((float)sE[row * 1024 + j] - mm);
    sE[row * 1024 + j] = (_Float16)e;                 // unnormalized attn
    s += e;
  }
  sStat[row * 8 + sub] = s;
  __syncthreads();
  if (sub == 0) {
    float ss = 0.f;
#pragma unroll
    for (int j = 0; j < 8; ++j) ss += sStat[row * 8 + j];
    sSum[row] = ss;
  }
  __syncthreads();

  // ---- attn @ V (each wave: its 256 kv, both 16-wide d tiles, B from LDS) ----
  wait_async0();  // V slab resident (per-wave region, per-wave wait)
  v8f acc0 = {}, acc1 = {};
#pragma unroll
  for (int kc = 0; kc < 8; ++kc) {
    v16h aA = a_from_f16(sE + l15 * 1024 + wvid * 256 + kc * 32, kkA);
    v16h b0 = b_from_f16(sKV + kc * 32 * 32 + l15, 32, kkB);
    v16h b1 = b_from_f16(sKV + kc * 32 * 32 + 16 + l15, 32, kkB);
    acc0 = wmma16(aA, b0, acc0);
    acc1 = wmma16(aA, b1, acc1);
  }
#pragma unroll
  for (int i = 0; i < 8; ++i) {
    int r = i + hi8;
    sPart[(wvid * 16 + r) * 32 + l15]      = acc0[i];
    sPart[(wvid * 16 + r) * 32 + 16 + l15] = acc1[i];
  }
  __syncthreads();
  for (int e = tid; e < 512; e += 128) {
    int r = e >> 5, d = e & 31;
    float s4 = sPart[r * 32 + d] + sPart[(16 + r) * 32 + d] +
               sPart[(32 + r) * 32 + d] + sPart[(48 + r) * 32 + d];
    float o = s4 / sSum[r];
    Ot[((size_t)(n * 256 + aH * 32 + d)) * HWQ + qt * 16 + r] = (_Float16)o;
  }
}

// ---------------- Kernel 7: output projection + residual ----------------
// Stages the shared 256x16 f16 Ot panel (8KB) per block via async LDS.
#define KPJ_SMEM 8192
__global__ void k_proj(const _Float16* __restrict__ Ot, const float* __restrict__ Wp,
                       const float* __restrict__ bp, const float* __restrict__ gamma,
                       const float* __restrict__ x, float* __restrict__ out) {
  extern __shared__ char dynp[];
  _Float16* sB = (_Float16*)dynp;               // [c 256][s 16]
  int tid = threadIdx.x, lane = tid & 31;
  int wave = blockIdx.x * 8 + (tid >> 5);       // 8192 waves
  int otile = wave & 15, stile = wave >> 4;     // 8 waves/block share stile
  int s0 = stile * 16;
  int n = s0 >> 12, hw0 = s0 & 4095;
  unsigned lbase = __builtin_amdgcn_groupstaticsize();
  {
    const _Float16* src = Ot + (size_t)n * CIN * HWQ + (size_t)tid * HWQ + hw0;
#pragma unroll
    for (int ch = 0; ch < 2; ++ch)              // row c=tid: 32B = 2 x b128
      async_ld_b128(lbase + (unsigned)(tid * 32 + ch * 16), src + ch * 8);
  }
  int kkA[8], kkB[8];
  lane_kkA(lane, kkA); lane_kkB(lane, kkB);
  int l15 = lane & 15, hi8 = (lane >> 4) << 3;
  wait_async0();
  __syncthreads();
  v8f acc = {};
#pragma unroll
  for (int kc = 0; kc < 8; ++kc) {
    v16h a = a_from_f32(Wp + (size_t)(otile * 16 + l15) * CIN + kc * 32, kkA);
    v16h b = b_from_f16(sB + kc * 32 * 16 + l15, 16, kkB);
    acc = wmma16(a, b, acc);
  }
  float g = gamma[0];
#pragma unroll
  for (int i = 0; i < 8; ++i) {
    int r = i + hi8;
    int o = otile * 16 + r;
    size_t idx = (size_t)n * CIN * HWQ + (size_t)o * HWQ + hw0 + l15;
    out[idx] = g * (acc[i] + bp[o]) + x[idx];
  }
}

// ---------------- launcher ----------------
extern "C" void kernel_launch(void* const* d_in, const int* in_sizes, int n_in,
                              void* d_out, int out_size, void* d_ws, size_t ws_size,
                              hipStream_t stream) {
  (void)in_sizes; (void)n_in; (void)out_size; (void)ws_size;
  const float* x     = (const float*)d_in[0];
  const float* Wq    = (const float*)d_in[1];
  const float* Wk    = (const float*)d_in[2];
  const float* Wv    = (const float*)d_in[3];
  const float* Wx    = (const float*)d_in[4];
  const float* Wy    = (const float*)d_in[5];
  const float* ab    = (const float*)d_in[6];
  const float* gb    = (const float*)d_in[7];
  const float* Wproj = (const float*)d_in[8];
  const float* bproj = (const float*)d_in[9];
  const float* gamma = (const float*)d_in[10];
  float* out = (float*)d_out;

  char* ws = (char*)d_ws;                       // ~34 MB total
  _Float16* Qapp = (_Float16*)(ws + 0);         //  4 MB
  _Float16* Qgeo = (_Float16*)(ws + 4194304);   //  4 MB
  _Float16* Kt   = (_Float16*)(ws + 8388608);   //  1 MB
  _Float16* Vt   = (_Float16*)(ws + 9437184);   //  1 MB
  _Float16* E    = (_Float16*)(ws + 10485760);  //  512 KB
  _Float16* pxB  = (_Float16*)(ws + 11010048);  //  1 MB
  _Float16* pyB  = (_Float16*)(ws + 12058624);  //  1 MB
  float*    GX   = (float*)   (ws + 13107200);  //  8 MB
  float*    GY   = (float*)   (ws + 21495808);  //  8 MB
  _Float16* Ot   = (_Float16*)(ws + 29884416);  //  4 MB

  k_embed<<<1024, 256, 0, stream>>>(E);
  k_q    <<<1024, 256, KQ_SMEM,  stream>>>(x, Wq, ab, gb, Qapp, Qgeo);
  k_kv   <<<256,  256, KKV_SMEM, stream>>>(x, Wk, Wv, Kt, Vt);
  k_pos  <<<256,  256, 0, stream>>>(Wx, Wy, E, pxB, pyB);
  k_gxgy <<<2048, 256, 0, stream>>>(Qgeo, pxB, pyB, GX, GY);
  k_attn <<<4096, 128, ATTN_SMEM, stream>>>(Qapp, Kt, Vt, GX, GY, Ot);
  k_proj <<<1024, 256, KPJ_SMEM, stream>>>(Ot, Wproj, bproj, gamma, x, out);
}
